// GNNProcessor_38783554683641
// MI455X (gfx1250) — compile-verified, matrix operand order checked by reading, not run on hardware
//
#include <hip/hip_runtime.h>
#include <hip/hip_bf16.h>
#include <math.h>

typedef __attribute__((ext_vector_type(16))) _Float16     v16h;
typedef __attribute__((ext_vector_type(8)))  float        v8f;
typedef __attribute__((ext_vector_type(4)))  unsigned int v4u;
typedef __attribute__((ext_vector_type(8)))  int          v8i;
typedef __attribute__((ext_vector_type(4)))  int          v4i;

#define N_NODES 50000
#define N_EDGES 800000
#define DIM     128
#define HEADS   16
#define CHAN    8
#define DHID    512
#define LN_EPS  1e-5f

// ---------------------------------------------------------------------------
// WMMA helper: D = A(16x32 f16) x B(32x16 f16) + C(16x16 f32)
// ---------------------------------------------------------------------------
__device__ __forceinline__ v8f wmma16(v16h a, v16h b, v8f c) {
  return __builtin_amdgcn_wmma_f32_16x16x32_f16(
      /*neg_a=*/false, a, /*neg_b=*/false, b,
      /*c_mod=*/(short)0, c, /*reuse_a=*/false, /*reuse_b=*/false);
}

__device__ __forceinline__ void atomicMaxFloat(float* addr, float val) {
  // ordered-int trick; m initialized to -inf (0xFF800000)
  if (val >= 0.0f) atomicMax((int*)addr, __float_as_int(val));
  else             atomicMin((unsigned int*)addr, __float_as_uint(val));
}

// ---------------------------------------------------------------------------
// Tensor Data Mover: 1-D DMA of `ndw` dwords global -> LDS (TENSORcnt).
// D# layout per CDNA5 ISA (group0 128b, group1 256b).
// ndw must fit in 16 bits (tile_dim0). 6-arg builtin (clang-23 / therock HIP).
// ---------------------------------------------------------------------------
__device__ __forceinline__ void tdm_load_to_lds(const void* gsrc, unsigned lds_off,
                                                unsigned ndw) {
  unsigned long long ga = (unsigned long long)gsrc;
  v4u g0;
  g0[0] = 1u;                                        // count=1, user mode
  g0[1] = lds_off;                                   // lds_addr (bytes)
  g0[2] = (unsigned)(ga & 0xFFFFFFFFu);              // global_addr[31:0]
  g0[3] = (unsigned)((ga >> 32) & 0x01FFFFFFu)       // global_addr[56:32]
          | (2u << 30);                              // type = 2 ("image")
  v8i g1;
  g1[0] = (int)(2u << 16);                           // data_size=2 (4 bytes)
  g1[1] = (int)((ndw & 0xFFFFu) << 16);              // tensor_dim0[15:0] @ bit48
  g1[2] = (int)((ndw >> 16) & 0xFFFFu) | (1 << 16);  // tensor_dim0[31:16], tensor_dim1=1
  g1[3] = (int)((ndw & 0xFFFFu) << 16);              // tile_dim0 @ bit112
  g1[4] = 0;                                         // tile_dim1=0, tile_dim2=0
  g1[5] = (int)ndw;                                  // tensor_dim0_stride[31:0]
  g1[6] = 0;
  g1[7] = 0;
  v4i z4 = {0, 0, 0, 0};                             // groups 2/3 unused (<=2D)
  v8i z8 = {0, 0, 0, 0, 0, 0, 0, 0};
  __builtin_amdgcn_tensor_load_to_lds(g0, g1, z4, z4, z8, 0);
}

// ---------------------------------------------------------------------------
// Pack weights: fp32 [K][Nout] -> f16 transposed [Nout][K]; concat q|k|v|s.
// ---------------------------------------------------------------------------
__global__ void pack_kernel(const float* __restrict__ wq, const float* __restrict__ wk,
                            const float* __restrict__ wv, const float* __restrict__ w_s,
                            const float* __restrict__ we, const float* __restrict__ wp,
                            const float* __restrict__ w1, const float* __restrict__ w2,
                            const float* __restrict__ bq, const float* __restrict__ bk,
                            const float* __restrict__ bv, const float* __restrict__ bs,
                            _Float16* __restrict__ wqkvs_t, _Float16* __restrict__ we_t,
                            _Float16* __restrict__ wp_t,    _Float16* __restrict__ w1_t,
                            _Float16* __restrict__ w2_t,    float* __restrict__ bqkvs) {
  int i = blockIdx.x * blockDim.x + threadIdx.x;
  if (i < 65536) {                       // wqkvs_t [512][128]
    int n = i >> 7, k = i & 127;
    const float* w = (n < 128) ? wq : (n < 256) ? wk : (n < 384) ? wv : w_s;
    wqkvs_t[i] = (_Float16)w[k * 128 + (n & 127)];
  } else if (i < 65536 + 16384) {        // we_t [128][128]
    int j = i - 65536; int n = j >> 7, k = j & 127;
    we_t[j] = (_Float16)we[k * 128 + n];
  } else if (i < 65536 + 32768) {        // wp_t [128][128]
    int j = i - 65536 - 16384; int n = j >> 7, k = j & 127;
    wp_t[j] = (_Float16)wp[k * 128 + n];
  } else if (i < 65536 + 32768 + 65536) { // w1_t [512][128]
    int j = i - 65536 - 32768; int n = j >> 7, k = j & 127;
    w1_t[j] = (_Float16)w1[k * 512 + n];
  } else if (i < 65536 + 32768 + 131072) { // w2_t [128][512]
    int j = i - 65536 - 32768 - 65536; int n = j >> 9, k = j & 511;
    w2_t[j] = (_Float16)w2[k * 128 + n];
  } else if (i < 65536 + 32768 + 131072 + 512) { // bqkvs [512]
    int j = i - (65536 + 32768 + 131072);
    bqkvs[j] = (j < 128) ? bq[j] : (j < 256) ? bk[j - 128]
             : (j < 384) ? bv[j - 256] : bs[j - 384];
  }
}

// ---------------------------------------------------------------------------
// Per-call init: m = -inf, denom = 0, agg = 0 (deterministic across replays)
// ---------------------------------------------------------------------------
__global__ void init_kernel(float* __restrict__ mbuf, float* __restrict__ denom,
                            float* __restrict__ agg) {
  int i = blockIdx.x * blockDim.x + threadIdx.x;
  if (i < N_NODES * HEADS) { mbuf[i] = -INFINITY; denom[i] = 0.0f; }
  if (i < N_NODES * DIM)   agg[i] = 0.0f;
}

// ---------------------------------------------------------------------------
// LayerNorm over 128 channels, one warp per row, f16 output for WMMA
// ---------------------------------------------------------------------------
__global__ void ln_kernel(const float* __restrict__ in, const float* __restrict__ g,
                          const float* __restrict__ b, _Float16* __restrict__ out,
                          int rows) {
  int warp = (blockIdx.x * blockDim.x + threadIdx.x) >> 5;
  int lane = threadIdx.x & 31;
  if (warp >= rows) return;
  const float* r = in + (size_t)warp * DIM;
  float v0 = r[lane], v1 = r[lane + 32], v2 = r[lane + 64], v3 = r[lane + 96];
  float s  = v0 + v1 + v2 + v3;
  float s2 = v0 * v0 + v1 * v1 + v2 * v2 + v3 * v3;
  #pragma unroll
  for (int off = 16; off > 0; off >>= 1) {
    s  += __shfl_xor(s, off);
    s2 += __shfl_xor(s2, off);
  }
  float mean = s * (1.0f / DIM);
  float var  = s2 * (1.0f / DIM) - mean * mean;
  float rs   = rsqrtf(var + LN_EPS);
  _Float16* o = out + (size_t)warp * DIM;
  o[lane]      = (_Float16)((v0 - mean) * rs * g[lane]      + b[lane]);
  o[lane + 32] = (_Float16)((v1 - mean) * rs * g[lane + 32] + b[lane + 32]);
  o[lane + 64] = (_Float16)((v2 - mean) * rs * g[lane + 64] + b[lane + 64]);
  o[lane + 96] = (_Float16)((v3 - mean) * rs * g[lane + 96] + b[lane + 96]);
}

// ---------------------------------------------------------------------------
// Generic WMMA GEMM: C[M,Dout] = A[M,K](f16) x Wt[Dout,K](f16) + bias (+res)
// One wave computes a 16-row strip. OUT_MODE 0: f32(+res); 1: f16 SiLU.
// ---------------------------------------------------------------------------
template <int K, bool CACHE_A, int OUT_MODE>
__global__ __launch_bounds__(256)
void gemm_wmma_kernel(const _Float16* __restrict__ A, const _Float16* __restrict__ Wt,
                      const float* __restrict__ bias, const float* __restrict__ res,
                      float* __restrict__ Cf, _Float16* __restrict__ Ch,
                      int M, int Dout) {
  const int wave = threadIdx.x >> 5;
  const int lane = threadIdx.x & 31;
  const int lr = lane & 15, hi = lane >> 4;
  const int rowTile = blockIdx.x * (blockDim.x >> 5) + wave;
  const int row0 = rowTile * 16;
  if (row0 >= M) return;   // wave-uniform: EXEC stays all-ones for WMMA

  const _Float16* Arow = A + (size_t)(row0 + lr) * K + hi * 16;
  v16h af[CACHE_A ? (K / 32) : 1];
  if (CACHE_A) {
    #pragma unroll
    for (int kk = 0; kk < K / 32; kk++) af[kk] = *(const v16h*)(Arow + kk * 32);
  }

  for (int nt = 0; nt < Dout / 16; nt++) {
    v8f acc = {};
    const _Float16* Brow = Wt + (size_t)(nt * 16 + lr) * K + hi * 16;
    #pragma unroll
    for (int kk = 0; kk < K / 32; kk++) {
      v16h bfr = *(const v16h*)(Brow + kk * 32);
      v16h afr = CACHE_A ? af[kk] : *(const v16h*)(Arow + kk * 32);
      acc = wmma16(afr, bfr, acc);
    }
    const int col = nt * 16 + lr;
    const float bvv = bias ? bias[col] : 0.0f;
    const int rbase = row0 + hi * 8;
    #pragma unroll
    for (int r = 0; r < 8; r++) {
      const size_t oi = (size_t)(rbase + r) * Dout + col;
      float v = acc[r] + bvv;
      if (OUT_MODE == 0) {
        if (res) v += res[oi];
        Cf[oi] = v;
      } else {
        Ch[oi] = (_Float16)(v / (1.0f + __expf(-v)));  // SiLU
      }
    }
  }
}

// ---------------------------------------------------------------------------
// Fused edge kernel: we_t staged into LDS via TDM (async, TENSORcnt), then
// eproj = edge_attr @ we + be via WMMA -> LDS, then
// alpha[e,h] = (q[dst] . (k[src] + eproj)) / sqrt(C), atomic segment max.
// Block = 128 threads (4 waves), 64 edges, 64KB LDS (32KB kj + 32KB weights).
// ---------------------------------------------------------------------------
__global__ __launch_bounds__(128)
void edge_attn_kernel(const float* __restrict__ edge_attr,
                      const _Float16* __restrict__ we_t,
                      const float* __restrict__ be,
                      const int* __restrict__ src, const int* __restrict__ dst,
                      const float* __restrict__ qkvs,
                      float* __restrict__ alpha, float* __restrict__ mbuf) {
  __shared__ float    kj[64 * DIM];        // 32KB: eproj tiles
  __shared__ _Float16 lds_we[DIM * DIM];   // 32KB: staged weight matrix
  const int tid = threadIdx.x;
  const int wave = tid >> 5, lane = tid & 31;
  const int lr = lane & 15, hi = lane >> 4;
  const int e0 = blockIdx.x * 64;

  // async stage we_t (8192 dwords) into LDS via the Tensor Data Mover;
  // wave 0 issues once (TDM ignores EXEC; one instruction per wave).
  if (wave == 0) {
    unsigned lds_off = (unsigned)(unsigned long long)(void*)lds_we;  // LDS aperture: low 32b = byte offset
    tdm_load_to_lds((const void*)we_t, lds_off, (DIM * DIM * 2) / 4);
    __builtin_amdgcn_s_wait_tensorcnt(0);
  }

  // overlap: build f16 A fragments (fp32->f16 convert) while TDM runs
  const int eBase = e0 + wave * 16;
  const float* arow = edge_attr + (size_t)(eBase + lr) * DIM + hi * 16;
  v16h af[4];
  #pragma unroll
  for (int kk = 0; kk < 4; kk++) {
    const float4* p = (const float4*)(arow + kk * 32);
    float4 f0 = p[0], f1 = p[1], f2 = p[2], f3 = p[3];
    v16h a;
    a[0]=(_Float16)f0.x;  a[1]=(_Float16)f0.y;  a[2]=(_Float16)f0.z;  a[3]=(_Float16)f0.w;
    a[4]=(_Float16)f1.x;  a[5]=(_Float16)f1.y;  a[6]=(_Float16)f1.z;  a[7]=(_Float16)f1.w;
    a[8]=(_Float16)f2.x;  a[9]=(_Float16)f2.y;  a[10]=(_Float16)f2.z; a[11]=(_Float16)f2.w;
    a[12]=(_Float16)f3.x; a[13]=(_Float16)f3.y; a[14]=(_Float16)f3.z; a[15]=(_Float16)f3.w;
    af[kk] = a;
  }
  __syncthreads();   // weights resident in LDS for all waves

  // phase 1: WMMA edge projection, B fragments from LDS
  #pragma unroll
  for (int nt = 0; nt < 8; nt++) {
    v8f acc = {};
    const _Float16* Brow = lds_we + (nt * 16 + lr) * DIM + hi * 16;
    #pragma unroll
    for (int kk = 0; kk < 4; kk++) {
      v16h bfr = *(const v16h*)(Brow + kk * 32);
      acc = wmma16(af[kk], bfr, acc);
    }
    const float bvv = be[nt * 16 + lr];
    #pragma unroll
    for (int r = 0; r < 8; r++)
      kj[(wave * 16 + hi * 8 + r) * DIM + nt * 16 + lr] = acc[r] + bvv;
  }
  __syncthreads();

  // phase 2: per-(edge,head) attention logits + segment max
  #pragma unroll
  for (int it = 0; it < 8; it++) {
    int idx = it * 128 + tid;          // [0, 64*16)
    int el = idx >> 4, hh = idx & 15;
    int e = e0 + el;
    int s = src[e], d = dst[e];
    const float* kp  = qkvs + (size_t)s * 512 + 128 + hh * CHAN;  // k[src]
    const float* qp  = qkvs + (size_t)d * 512 +       hh * CHAN;  // q[dst]
    const float* kjl = kj + el * DIM + hh * CHAN;
    float dot = 0.0f;
    #pragma unroll
    for (int c = 0; c < CHAN; c++) dot += (kjl[c] + kp[c]) * qp[c];
    float a = dot * 0.35355339059327379f;  // 1/sqrt(8)
    alpha[(size_t)e * HEADS + hh] = a;
    atomicMaxFloat(&mbuf[(size_t)d * HEADS + hh], a);
  }
}

// ---------------------------------------------------------------------------
// Softmax pass 2: ex = exp(alpha - m[dst]); denom += ex  (in-place on alpha)
// ---------------------------------------------------------------------------
__global__ void expden_kernel(float* __restrict__ alpha, const int* __restrict__ dst,
                              const float* __restrict__ mbuf, float* __restrict__ denom) {
  int i = blockIdx.x * blockDim.x + threadIdx.x;
  if (i >= N_EDGES * HEADS) return;
  int e = i >> 4, h = i & 15;
  int d = dst[e];
  float ex = __expf(alpha[i] - mbuf[d * HEADS + h]);
  alpha[i] = ex;
  atomicAdd(&denom[d * HEADS + h], ex);
}

// ---------------------------------------------------------------------------
// Weighted scatter-add: agg[dst,h,:] += v[src,h,:] * ex/denom[dst,h]
// ---------------------------------------------------------------------------
__global__ void scatter_kernel(const float* __restrict__ alpha, const int* __restrict__ src,
                               const int* __restrict__ dst, const float* __restrict__ denom,
                               const float* __restrict__ qkvs, float* __restrict__ agg) {
  int i = blockIdx.x * blockDim.x + threadIdx.x;
  if (i >= N_EDGES * HEADS) return;
  int e = i >> 4, h = i & 15;
  int s = src[e], d = dst[e];
  float w = alpha[i] / denom[d * HEADS + h];
  const float* vp = qkvs + (size_t)s * 512 + 256 + h * CHAN;   // v[src]
  float* ap = agg + (size_t)d * DIM + h * CHAN;
  #pragma unroll
  for (int c = 0; c < CHAN; c++) atomicAdd(&ap[c], vp[c] * w);
}

// s16 = f16(agg + x_r)  (x_r is the 4th slot of qkvs)
__global__ void make_s16_kernel(const float* __restrict__ agg,
                                const float* __restrict__ qkvs,
                                _Float16* __restrict__ s16) {
  int i = blockIdx.x * blockDim.x + threadIdx.x;
  if (i >= N_NODES * DIM) return;
  int n = i >> 7, c = i & 127;
  s16[i] = (_Float16)(agg[i] + qkvs[(size_t)n * 512 + 384 + c]);
}

// ---------------------------------------------------------------------------
extern "C" void kernel_launch(void* const* d_in, const int* in_sizes, int n_in,
                              void* d_out, int out_size, void* d_ws, size_t ws_size,
                              hipStream_t stream) {
  const float* x         = (const float*)d_in[0];
  const int*   eidx      = (const int*)  d_in[1];
  const float* edge_attr = (const float*)d_in[2];
  const float* ln1_g     = (const float*)d_in[3];
  const float* ln1_b     = (const float*)d_in[4];
  const float* wq = (const float*)d_in[5];  const float* bq = (const float*)d_in[6];
  const float* wk = (const float*)d_in[7];  const float* bk = (const float*)d_in[8];
  const float* wv = (const float*)d_in[9];  const float* bv = (const float*)d_in[10];
  const float* w_s = (const float*)d_in[11]; const float* bs = (const float*)d_in[12];
  const float* we = (const float*)d_in[13]; const float* be = (const float*)d_in[14];
  const float* wp = (const float*)d_in[15]; const float* bp = (const float*)d_in[16];
  const float* mlp_g = (const float*)d_in[17];
  const float* mlp_b = (const float*)d_in[18];
  const float* w1 = (const float*)d_in[19]; const float* b1 = (const float*)d_in[20];
  const float* w2 = (const float*)d_in[21]; const float* b2 = (const float*)d_in[22];
  const int* src = eidx;
  const int* dst = eidx + N_EDGES;

  // workspace layout (bytes)
  char* wsp = (char*)d_ws;
  const size_t OFS_QKVS  = 0;                               // N*512*4 = 102.4MB (reused for h1)
  const size_t OFS_H16   = OFS_QKVS + (size_t)N_NODES*512*4; // N*128*2 (h16 / s16 / ln16)
  const size_t OFS_ALPHA = OFS_H16  + (size_t)N_NODES*DIM*2; // E*16*4
  const size_t OFS_M     = OFS_ALPHA+ (size_t)N_EDGES*HEADS*4;
  const size_t OFS_DEN   = OFS_M    + (size_t)N_NODES*HEADS*4;
  const size_t OFS_AGG   = OFS_DEN  + (size_t)N_NODES*HEADS*4;
  const size_t OFS_OUT   = OFS_AGG  + (size_t)N_NODES*DIM*4;
  const size_t OFS_W     = OFS_OUT  + (size_t)N_NODES*DIM*4;

  float*    qkvs    = (float*)   (wsp + OFS_QKVS);
  _Float16* h16     = (_Float16*)(wsp + OFS_H16);   // also s16 / ln16 (sequential reuse)
  float*    alpha   = (float*)   (wsp + OFS_ALPHA);
  float*    mbuf    = (float*)   (wsp + OFS_M);
  float*    denom   = (float*)   (wsp + OFS_DEN);
  float*    agg     = (float*)   (wsp + OFS_AGG);
  float*    outbuf  = (float*)   (wsp + OFS_OUT);
  _Float16* wqkvs_t = (_Float16*)(wsp + OFS_W);
  _Float16* we_t    = wqkvs_t + 512 * 128;
  _Float16* wp_t    = we_t    + 128 * 128;
  _Float16* w1_t    = wp_t    + 128 * 128;
  _Float16* w2_t    = w1_t    + 512 * 128;
  float*    bqkvs   = (float*)(w2_t + 128 * 512);
  _Float16* h1      = (_Float16*)(wsp + OFS_QKVS);  // reuse after scatter/make_s16

  // 1. pack weights to f16 transposed
  pack_kernel<<<(229888 + 255) / 256, 256, 0, stream>>>(
      wq, wk, wv, w_s, we, wp, w1, w2, bq, bk, bv, bs,
      wqkvs_t, we_t, wp_t, w1_t, w2_t, bqkvs);
  // 2. init segment buffers
  init_kernel<<<(N_NODES * DIM + 255) / 256, 256, 0, stream>>>(mbuf, denom, agg);
  // 3. LN1 -> h16
  ln_kernel<<<(N_NODES * 32 + 255) / 256, 256, 0, stream>>>(x, ln1_g, ln1_b, h16, N_NODES);
  // 4. fused q|k|v|x_r projection: [N,128] x [128,512]
  gemm_wmma_kernel<128, true, 0><<<(N_NODES + 127) / 128, 256, 0, stream>>>(
      h16, wqkvs_t, bqkvs, nullptr, qkvs, nullptr, N_NODES, 512);
  // 5. fused edge projection (TDM-staged weights) + attention logits + segment max
  edge_attn_kernel<<<N_EDGES / 64, 128, 0, stream>>>(
      edge_attr, we_t, be, src, dst, qkvs, alpha, mbuf);
  // 6. exp + denom
  expden_kernel<<<(N_EDGES * HEADS + 255) / 256, 256, 0, stream>>>(alpha, dst, mbuf, denom);
  // 7. weighted scatter-add aggregation
  scatter_kernel<<<(N_EDGES * HEADS + 255) / 256, 256, 0, stream>>>(
      alpha, src, dst, denom, qkvs, agg);
  // 8. s16 = f16(agg + x_r)
  make_s16_kernel<<<(N_NODES * DIM + 255) / 256, 256, 0, stream>>>(agg, qkvs, h16);
  // 9. out = s16 @ wp + bp + x
  gemm_wmma_kernel<128, true, 0><<<(N_NODES + 127) / 128, 256, 0, stream>>>(
      h16, wp_t, bp, x, outbuf, nullptr, N_NODES, 128);
  // 10. LN2 -> ln16 (reuses h16 buffer)
  ln_kernel<<<(N_NODES * 32 + 255) / 256, 256, 0, stream>>>(outbuf, mlp_g, mlp_b, h16, N_NODES);
  // 11. h1 = f16(silu(ln16 @ w1 + b1))
  gemm_wmma_kernel<128, true, 1><<<(N_NODES + 127) / 128, 256, 0, stream>>>(
      h16, w1_t, b1, nullptr, nullptr, h1, N_NODES, 512);
  // 12. d_out = h1 @ w2 + b2 + out
  gemm_wmma_kernel<512, false, 0><<<(N_NODES + 127) / 128, 256, 0, stream>>>(
      h1, w2_t, b2, outbuf, (float*)d_out, nullptr, N_NODES, 128);
}
